// CapsuleNetwork_5007931867312
// MI455X (gfx1250) — compile-verified, matrix operand order checked by reading, not run on hardware
//
#include <hip/hip_runtime.h>
#include <hip/hip_bf16.h>

// CapsNet forward for gfx1250 (MI455X).
// conv2 (196 GFLOP) + decoder GEMMs run through v_wmma_f32_16x16x32_bf16.
// conv2 staging now uses CDNA5 async global->LDS copies (ASYNCcnt),
// triple-buffered LDS, one barrier per K-step.

typedef __attribute__((ext_vector_type(16))) __bf16 v16bf;
typedef __attribute__((ext_vector_type(8)))  float  v8f;

#define BATCH 512
#define NPRIM 1152
#define NSEC  10

// ---------- helpers ----------
__device__ __forceinline__ unsigned short f2bf(float f) {
  union { float f; unsigned int u; } x; x.f = f;
  unsigned int r = x.u + 0x7FFFu + ((x.u >> 16) & 1u);   // RNE
  return (unsigned short)(r >> 16);
}

// Fragment load from an LDS tile stored row-major with stride 40 elements.
// 16-bit A/B layout (ISA 7.12.2): lane L -> row L%16; K chunk base kb=(L<16?0:8),
// holding k = kb..kb+7 and kb+16..kb+23  (two contiguous 16B reads).
__device__ __forceinline__ v16bf ldfrag(const unsigned short* s, int rowBase, int lane) {
  int r  = rowBase + (lane & 15);
  int kb = (lane & 16) ? 8 : 0;
  const unsigned short* p = s + r * 40 + kb;
  union { uint4 q[2]; v16bf v; } u;
  u.q[0] = *(const uint4*)(p);
  u.q[1] = *(const uint4*)(p + 16);
  return u.v;
}

__device__ __forceinline__ v8f wmma_bf16(v16bf a, v16bf b, v8f c) {
  return __builtin_amdgcn_wmma_f32_16x16x32_bf16(false, a, false, b, (short)0, c, false, false);
}

// Async global->LDS copy of 32 contiguous bytes per lane (2 x b128).
// INST_OFFSET applies to both the LDS and global address (ISA 08 §4.4).
__device__ __forceinline__ void async_copy32(unsigned lds, const unsigned short* g) {
  asm volatile("global_load_async_to_lds_b128 %0, %1, off\n\t"
               "global_load_async_to_lds_b128 %0, %1, off offset:16"
               :: "v"(lds), "v"(g) : "memory");
}

// ---------- conv1: 9x9, 1->256, stride1, relu, output bf16 NHWC (512,20,20,256) ----------
__global__ __launch_bounds__(256) void conv1_kernel(
    const float* __restrict__ x, const float* __restrict__ k,
    const float* __restrict__ bias, unsigned short* __restrict__ h) {
  int bid = blockIdx.x;            // b*400 + oy*20 + ox
  int co  = threadIdx.x;
  int b   = bid / 400;
  int rem = bid - b * 400;
  int oy  = rem / 20, ox = rem - oy * 20;
  const float* xp = x + ((size_t)b * 28 + oy) * 28 + ox;
  float acc = bias[co];
  for (int ky = 0; ky < 9; ++ky)
    for (int kx = 0; kx < 9; ++kx)
      acc += xp[ky * 28 + kx] * k[(ky * 9 + kx) * 256 + co];
  acc = fmaxf(acc, 0.f);
  h[(size_t)bid * 256 + co] = f2bf(acc);
}

// ---------- conv2 weights: (9,9,256,256) f32 -> transposed bf16 W2T[co][k], k=tap*256+ci ----------
__global__ __launch_bounds__(256) void convert_w2(
    const float* __restrict__ k2, unsigned short* __restrict__ w2t) {
  size_t id = (size_t)blockIdx.x * 256 + threadIdx.x;   // 256*20736 total
  int kk = (int)(id % 20736);
  int n  = (int)(id / 20736);
  w2t[id] = f2bf(k2[(size_t)kk * 256 + n]);
}

// ---------- conv2 as implicit GEMM: (18432 x 20736) x (20736 x 256), bf16 WMMA ----------
// Async-staged, triple-buffered: issue step s+1, wait own asynccnt<=4 (step s landed),
// single barrier, compute step s. Buffer (s+1)%3 was last read at step s-2, protected
// by the single intervening barrier (step s-1).
__device__ __forceinline__ void conv2_stage(
    const unsigned short* h, const unsigned short* w2t,
    int rowBase, size_t bRow, int m, int half,
    unsigned short (*As)[128 * 40], unsigned short (*Bs)[128 * 40],
    int step, int buf) {
  int k0  = step << 5;
  int tap = k0 >> 8, c0 = k0 & 255;
  int ky  = tap / 9, kx = tap - ky * 9;
  const unsigned short* ap = h + rowBase + (ky * 20 + kx) * 256 + c0 + half * 16;
  const unsigned short* bp = w2t + bRow + k0 + half * 16;
  unsigned la = (unsigned)(size_t)&As[buf][m * 40 + half * 16];  // low 32 bits = LDS offset
  unsigned lb = (unsigned)(size_t)&Bs[buf][m * 40 + half * 16];
  async_copy32(la, ap);
  async_copy32(lb, bp);
}

__global__ __launch_bounds__(256) void conv2_wmma(
    const unsigned short* __restrict__ h, const unsigned short* __restrict__ w2t,
    const float* __restrict__ bias, float* __restrict__ u) {
  __shared__ __align__(16) unsigned short As[3][128 * 40];
  __shared__ __align__(16) unsigned short Bs[3][128 * 40];
  int t = threadIdx.x, lane = t & 31, wave = t >> 5;
  int wm = wave >> 1, wn = wave & 1;           // 4x2 wave grid; wave owns 32x64
  int m0 = blockIdx.x * 128, n0 = blockIdx.y * 128;

  v8f zero = {0.f,0.f,0.f,0.f,0.f,0.f,0.f,0.f};
  v8f acc[2][4];
  for (int i = 0; i < 2; ++i) for (int j = 0; j < 4; ++j) acc[i][j] = zero;

  // staging roles: thread -> (row m = t/2, half = t&1); each half copies 32B
  int m = t >> 1, half = t & 1;
  int gm = m0 + m;
  int bb = gm / 36, pix = gm - bb * 36;
  int oy = pix / 6, ox = pix - oy * 6;
  const int rowBase = ((bb * 20 + oy * 2) * 20 + ox * 2) * 256;  // + (ky*20+kx)*256 + c
  const size_t bRow = (size_t)(n0 + m) * 20736;

  conv2_stage(h, w2t, rowBase, bRow, m, half, As, Bs, 0, 0);

  for (int s = 0; s < 648; ++s) {
    if (s + 1 < 648) {
      conv2_stage(h, w2t, rowBase, bRow, m, half, As, Bs, s + 1, (s + 1) % 3);
      asm volatile("s_wait_asynccnt 4" ::: "memory");   // step s's copies landed
    } else {
      asm volatile("s_wait_asynccnt 0" ::: "memory");
    }
    __syncthreads();

    const unsigned short* as = As[s % 3];
    const unsigned short* bs = Bs[s % 3];
    v16bf bf[4], af[2];
    for (int ni = 0; ni < 4; ++ni) bf[ni] = ldfrag(bs, wn * 64 + ni * 16, lane);
    for (int mi = 0; mi < 2; ++mi) af[mi] = ldfrag(as, wm * 32 + mi * 16, lane);
    for (int mi = 0; mi < 2; ++mi)
      for (int ni = 0; ni < 4; ++ni)
        acc[mi][ni] = wmma_bf16(af[mi], bf[ni], acc[mi][ni]);
  }

  int rlo  = (lane < 16) ? 0 : 8;          // D layout: VGPR j -> row j or j+8
  int col0 = n0 + wn * 64 + (lane & 15);
  for (int mi = 0; mi < 2; ++mi)
    for (int ni = 0; ni < 4; ++ni) {
      int col = col0 + ni * 16;
      float bv = bias[col];
      for (int j = 0; j < 8; ++j) {
        int row = m0 + wm * 32 + mi * 16 + rlo + j;
        u[(size_t)row * 256 + col] = acc[mi][ni][j] + bv;
      }
    }
}

// ---------- u_hat[b,n,s,d] = sum_p w[n,s,d,p] * u[b,n,p]   (K=8, VALU) ----------
__global__ __launch_bounds__(256) void uhat_kernel(
    const float* __restrict__ w, const float* __restrict__ u, float* __restrict__ uhat) {
  size_t id = (size_t)blockIdx.x * 256 + threadIdx.x;  // 512*1152*160 total
  int sd = (int)(id % 160);
  size_t bn = id / 160;
  int n = (int)(bn % NPRIM);
  const float* wp = w + ((size_t)n * 160 + sd) * 8;
  const float* up = u + bn * 8;
  float acc = 0.f;
#pragma unroll
  for (int p = 0; p < 8; ++p) acc += wp[p] * up[p];
  uhat[id] = acc;
}

// ---------- dynamic routing (3 iters), squash, mask; one block per batch elem ----------
__global__ __launch_bounds__(256) void routing_kernel(
    const float* __restrict__ uhat, const int* __restrict__ labels,
    float* __restrict__ v_out, float* __restrict__ vm_out) {
  int b = blockIdx.x, t = threadIdx.x;
  __shared__ float blog[NPRIM * NSEC];     // 46080 B
  __shared__ float nmax[NPRIM], nden[NPRIM];
  __shared__ float svec[160], vloc[160];
  const float* U = uhat + (size_t)b * NPRIM * 160;

  for (int i = t; i < NPRIM * NSEC; i += 256) blog[i] = 0.f;
  __syncthreads();

  for (int it = 0; it < 3; ++it) {
    for (int n = t; n < NPRIM; n += 256) {
      float mx = -1e30f;
      for (int s = 0; s < NSEC; ++s) mx = fmaxf(mx, blog[n * NSEC + s]);
      float den = 0.f;
      for (int s = 0; s < NSEC; ++s) den += __expf(blog[n * NSEC + s] - mx);
      nmax[n] = mx; nden[n] = den;
    }
    if (t < 160) svec[t] = 0.f;
    __syncthreads();

    for (int idx = t; idx < NPRIM * 160; idx += 256) {
      int n = idx / 160, sd = idx - n * 160, s = sd >> 4;
      float c = __expf(blog[n * NSEC + s] - nmax[n]) / nden[n];
      atomicAdd(&svec[sd], c * U[idx]);
    }
    __syncthreads();

    if (t < NSEC) {
      float sq = 0.f;
      for (int d = 0; d < 16; ++d) sq += svec[t * 16 + d] * svec[t * 16 + d];
      float nrm = sqrtf(sq);
      float sc = sq / (1.f + sq) / (nrm + 1e-7f);
      for (int d = 0; d < 16; ++d) vloc[t * 16 + d] = svec[t * 16 + d] * sc;
    }
    __syncthreads();

    if (it < 2) {
      for (int idx = t; idx < NPRIM * NSEC; idx += 256) {
        int n = idx / NSEC, s = idx - n * NSEC;
        const float* up = U + n * 160 + s * 16;
        float dot = 0.f;
        for (int d = 0; d < 16; ++d) dot += up[d] * vloc[s * 16 + d];
        blog[idx] += dot;
      }
      __syncthreads();
    }
  }

  if (t < 160) {
    v_out[(size_t)b * 160 + t] = vloc[t];
    int lab = labels[b];
    vm_out[(size_t)b * 160 + t] = ((t >> 4) == lab) ? vloc[t] : 0.f;
  }
}

// ---------- generic GEMM (f32 in/out, bf16 WMMA), bias + act (0 none,1 relu,2 sigmoid) ----------
__global__ __launch_bounds__(256) void gemm_wmma(
    const float* __restrict__ A, const float* __restrict__ B,
    const float* __restrict__ bias, float* __restrict__ C,
    int M, int N, int K, int act) {
  __shared__ __align__(16) unsigned short As[128 * 40];
  __shared__ __align__(16) unsigned short Bs[128 * 40];   // transposed: [n_local][k_local]
  int t = threadIdx.x, lane = t & 31, wave = t >> 5;
  int wm = wave >> 1, wn = wave & 1;
  int m0 = blockIdx.x * 128, n0 = blockIdx.y * 128;

  v8f zero = {0.f,0.f,0.f,0.f,0.f,0.f,0.f,0.f};
  v8f acc[2][4];
  for (int i = 0; i < 2; ++i) for (int j = 0; j < 4; ++j) acc[i][j] = zero;

  int m = t >> 1, half = t & 1;          // A staging role
  int kt = t >> 3, j8 = t & 7;           // B staging role (k row, 16-col group)

  for (int k0 = 0; k0 < K; k0 += 32) {
    unsigned short tmpA[16];
    int gm = m0 + m;
    const float* ap = A + (size_t)gm * K + k0 + half * 16;
    bool mok = gm < M;
#pragma unroll
    for (int i = 0; i < 16; ++i) {
      int kk = k0 + half * 16 + i;
      tmpA[i] = (mok && kk < K) ? f2bf(ap[i]) : (unsigned short)0;
    }
    unsigned short tmpB[16];
    int krow = k0 + kt;
    const float* bp = B + (size_t)krow * N + n0 + j8 * 16;
    bool kok = krow < K;
#pragma unroll
    for (int i = 0; i < 16; ++i) {
      int n = n0 + j8 * 16 + i;
      tmpB[i] = (kok && n < N) ? f2bf(bp[i]) : (unsigned short)0;
    }
    __syncthreads();
    *(uint4*)&As[m * 40 + half * 16]     = *(uint4*)&tmpA[0];
    *(uint4*)&As[m * 40 + half * 16 + 8] = *(uint4*)&tmpA[8];
#pragma unroll
    for (int i = 0; i < 16; ++i) Bs[(j8 * 16 + i) * 40 + kt] = tmpB[i];
    __syncthreads();

    v16bf bf[4], af[2];
    for (int ni = 0; ni < 4; ++ni) bf[ni] = ldfrag(Bs, wn * 64 + ni * 16, lane);
    for (int mi = 0; mi < 2; ++mi) af[mi] = ldfrag(As, wm * 32 + mi * 16, lane);
    for (int mi = 0; mi < 2; ++mi)
      for (int ni = 0; ni < 4; ++ni)
        acc[mi][ni] = wmma_bf16(af[mi], bf[ni], acc[mi][ni]);
  }

  int rlo  = (lane < 16) ? 0 : 8;
  int col0 = n0 + wn * 64 + (lane & 15);
  for (int mi = 0; mi < 2; ++mi)
    for (int ni = 0; ni < 4; ++ni) {
      int col = col0 + ni * 16;
      if (col >= N) continue;
      float bv = bias[col];
      for (int j = 0; j < 8; ++j) {
        int row = m0 + wm * 32 + mi * 16 + rlo + j;
        if (row >= M) continue;
        float r = acc[mi][ni][j] + bv;
        if (act == 1) r = fmaxf(r, 0.f);
        else if (act == 2) r = 1.f / (1.f + __expf(-r));
        C[(size_t)row * N + col] = r;
      }
    }
}

// ---------- workspace layout (bytes) ----------
#define OFF_H    ((size_t)0)                       // h   bf16  512*20*20*256  = 104857600 B
#define OFF_W2T  ((size_t)104857600)               // w2t bf16  256*20736      =  10616832 B
#define OFF_U    ((size_t)115474432)               // u   f32   512*9216       =  18874368 B
#define OFF_UHAT ((size_t)134348800)               // uhat f32  512*1152*160   = 377487360 B
#define OFF_VM   ((size_t)511836160)               // vm  f32   512*160        =    327680 B
#define OFF_R1   ((size_t)512163840)               // r1  f32   512*512        =   1048576 B
#define OFF_R2   ((size_t)513212416)               // r2  f32   512*1024       =   2097152 B

extern "C" void kernel_launch(void* const* d_in, const int* in_sizes, int n_in,
                              void* d_out, int out_size, void* d_ws, size_t ws_size,
                              hipStream_t stream) {
  const float* x      = (const float*)d_in[0];
  const int*   labels = (const int*)  d_in[1];
  const float* c1k    = (const float*)d_in[2];
  const float* c1b    = (const float*)d_in[3];
  const float* c2k    = (const float*)d_in[4];
  const float* c2b    = (const float*)d_in[5];
  const float* w      = (const float*)d_in[6];
  const float* d1w    = (const float*)d_in[7];
  const float* d1b    = (const float*)d_in[8];
  const float* d2w    = (const float*)d_in[9];
  const float* d2b    = (const float*)d_in[10];
  const float* d3w    = (const float*)d_in[11];
  const float* d3b    = (const float*)d_in[12];

  char* ws = (char*)d_ws;
  unsigned short* h    = (unsigned short*)(ws + OFF_H);
  unsigned short* w2t  = (unsigned short*)(ws + OFF_W2T);
  float* u    = (float*)(ws + OFF_U);
  float* uhat = (float*)(ws + OFF_UHAT);
  float* vm   = (float*)(ws + OFF_VM);
  float* r1   = (float*)(ws + OFF_R1);
  float* r2   = (float*)(ws + OFF_R2);

  float* out_v     = (float*)d_out;            // (512,1,10,16) = 81920
  float* out_recon = (float*)d_out + 81920;    // (512,784)

  conv1_kernel   <<<512 * 400, 256, 0, stream>>>(x, c1k, c1b, h);
  convert_w2     <<<20736,     256, 0, stream>>>(c2k, w2t);
  conv2_wmma     <<<dim3(144, 2), 256, 0, stream>>>(h, w2t, c2b, u);
  uhat_kernel    <<<368640,    256, 0, stream>>>(w, u, uhat);
  routing_kernel <<<512,       256, 0, stream>>>(uhat, labels, out_v, vm);
  gemm_wmma      <<<dim3(4, 4), 256, 0, stream>>>(vm, d1w, d1b, r1, 512, 512, 160, 1);
  gemm_wmma      <<<dim3(4, 8), 256, 0, stream>>>(r1, d2w, d2b, r2, 512, 1024, 512, 1);
  gemm_wmma      <<<dim3(4, 7), 256, 0, stream>>>(r2, d3w, d3b, out_recon, 512, 784, 1024, 2);
}
